// InteractionUnit_62955630625084
// MI455X (gfx1250) — matrix operand order repaired and checked by reading, not compiled, not run on hardware
//
#include <hip/hip_runtime.h>
#include <hip/hip_bf16.h>

#define DEVINL static __device__ __forceinline__

typedef __attribute__((ext_vector_type(16))) __bf16 v16bf;
typedef __attribute__((ext_vector_type(8)))  float  v8f;
typedef __attribute__((ext_vector_type(4)))  int    v4i;
typedef __attribute__((address_space(1))) v4i as1_v4i;
typedef __attribute__((address_space(3))) v4i as3_v4i;

union FragAB {
  v16bf bf;
  unsigned int u32[8];
  unsigned short u16[16];
};

#define Nn 32
#define Mm 20
#define Cc 512
#define Ss 196      // T*H*W = 4*7*7
#define Ll 4
#define Pp 5        // Mm / Ll
#define KL 3920     // Mm * Ss
#define SP 208      // padded S tile (13*16)
#define ATTN_SCALE 0.04419417382415922f  // 512^-0.5

// ---- CDNA5 async global->LDS path (guarded; falls back to sync copy) ------
#if defined(__has_builtin)
#if __has_builtin(__builtin_amdgcn_global_load_async_to_lds_b128)
#define HAS_ASYNC_LDS 1
#endif
#endif

#ifdef HAS_ASYNC_LDS
#if __has_builtin(__builtin_amdgcn_s_wait_asynccnt)
#define WAIT_ASYNC(n) __builtin_amdgcn_s_wait_asynccnt(n)
#else
#define WAIT_ASYNC(n) asm volatile("s_wait_asynccnt %0" ::"n"(n) : "memory")
#endif
#else
#define WAIT_ASYNC(n) do {} while (0)
#endif

DEVINL void async_cp16(const void* g, void* l) {
#ifdef HAS_ASYNC_LDS
  __builtin_amdgcn_global_load_async_to_lds_b128((as1_v4i*)g, (as3_v4i*)l, 0, 0);
#else
  *(uint4*)l = *(const uint4*)g;
#endif
}

DEVINL unsigned short f2bf(float f) {
  unsigned int u = __float_as_uint(f);
  u += 0x7fffu + ((u >> 16) & 1u);
  return (unsigned short)(u >> 16);
}

DEVINL unsigned int pack2bf(float lo, float hi) {
  return (unsigned int)f2bf(lo) | ((unsigned int)f2bf(hi) << 16);
}

DEVINL v8f zero8() {
  v8f z = {0.f, 0.f, 0.f, 0.f, 0.f, 0.f, 0.f, 0.f};
  return z;
}

// ---------------------------------------------------------------------------
// Projection GEMM:  D[o, col] = sum_c W[o,c] * X[c, col] + bias(o)  (bf16 out)
// MODE 0: Q (person)              -> qf [N][C][S]   c-major
// MODE 1: K (others, + pos_k[m/5])-> kf [N][C][KL]  c-major
// MODE 2: V (others, + pos_v[m/5])-> vf [N][C][KL]  c-major
// ---------------------------------------------------------------------------
template <int MODE>
__global__ __launch_bounds__(256) void proj_kernel(
    const float* __restrict__ X, const float* __restrict__ Wt,
    const float* __restrict__ bias, const float* __restrict__ pos,
    unsigned short* __restrict__ out) {
  __shared__ unsigned short Ws[128][36];  // [o][c]
  __shared__ unsigned short Xs[SP][36];   // [s][c]

  const int tid = threadIdx.x;
  const int lane = tid & 31;
  const int wv = tid >> 5;

  const int blk = blockIdx.x;
  const int oBase = blockIdx.y * 128;
  int n, m;
  const float* xin;
  if (MODE == 0) {
    n = blk; m = 0;
    xin = X + (size_t)n * Cc * Ss;
  } else {
    n = blk / Mm; m = blk % Mm;
    xin = X + (size_t)blk * Cc * Ss;
  }
  const int l = m / Pp;

  v8f acc[13];
#pragma unroll
  for (int i = 0; i < 13; ++i) acc[i] = zero8();

  const int am = lane & 15;
  const int akh = (lane >> 4) * 8;
  const int bn = lane & 15;
  const int bkh = (lane >> 4) * 16;

  for (int c0 = 0; c0 < Cc; c0 += 32) {
    __syncthreads();
#pragma unroll
    for (int i = 0; i < 16; ++i) {  // W tile 128x32, c fast
      int e = tid + i * 256;
      int c = e & 31, o = e >> 5;
      Ws[o][c] = f2bf(Wt[(size_t)(oBase + o) * Cc + c0 + c]);
    }
#pragma unroll
    for (int i = 0; i < 26; ++i) {  // X tile [208 s][32 c], s fast
      int e = tid + i * 256;
      int s = e % SP, c = e / SP;
      float v = (s < Ss) ? xin[(size_t)(c0 + c) * Ss + s] : 0.f;
      Xs[s][c] = f2bf(v);
    }
    __syncthreads();

    FragAB a;
#pragma unroll
    for (int j = 0; j < 8; ++j) {
      int kp = 2 * akh + 2 * j;
      a.u32[j] = *(const unsigned int*)&Ws[wv * 16 + am][kp];
    }
#pragma unroll
    for (int st = 0; st < 13; ++st) {
      FragAB b;
#pragma unroll
      for (int j = 0; j < 8; ++j)
        b.u32[j] = *(const unsigned int*)&Xs[st * 16 + bn][bkh + 2 * j];
      acc[st] = __builtin_amdgcn_wmma_f32_16x16x32_bf16(
          false, a.bf, false, b.bf, (short)0, acc[st], false, false);
    }
  }

  const size_t lenTot = (MODE == 0) ? (size_t)Ss : (size_t)KL;
  const size_t colBase = (MODE == 0) ? 0 : (size_t)m * Ss;
  unsigned short* obase = out + (size_t)n * Cc * lenTot;
#pragma unroll
  for (int st = 0; st < 13; ++st) {
    int s = st * 16 + (lane & 15);
    if (s >= Ss) continue;
#pragma unroll
    for (int r = 0; r < 8; ++r) {
      int o = oBase + wv * 16 + r + 8 * (lane >> 4);
      float bval = bias[o];
      if (MODE != 0) bval += pos[l * Cc + o];
      obase[(size_t)o * lenTot + colBase + s] = f2bf(acc[st][r] + bval);
    }
  }
}

// ---------------------------------------------------------------------------
// Flash attention: 32 queries/WG, 128-key blocks, full c=512, async
// double-buffered V stream. grid (7 qblocks, 32 samples), block 256.
// ---------------------------------------------------------------------------
__global__ __launch_bounds__(256) void attn_kernel(
    const unsigned short* __restrict__ qf,  // [N][C][S]
    const unsigned short* __restrict__ kf,  // [N][C][KL]
    const unsigned short* __restrict__ vf,  // [N][C][KL]
    float* __restrict__ ow) {               // [N][S][C]
  __shared__ unsigned short Qs[32][Cc + 4];    // 33 KB
  __shared__ unsigned short Ks[128][Cc + 4];   // 129 KB
  __shared__ unsigned short Vs[2][Cc][40];     // 80 KB (double-buffered 32-key chunks)
  __shared__ float Ls[32][132];                // 16.9 KB (pad: <=2-way banks)
  __shared__ unsigned short Ps[32][136];       // 8.7 KB (row = 272B, 16B aligned)
  __shared__ float rowM[32], rowS[32], rowScale[32];

  const int tid = threadIdx.x, lane = tid & 31, wv = tid >> 5;
  const int n = blockIdx.y, sBase = blockIdx.x * 32;
  const int qt = wv & 1;
  const int kc = wv >> 1;

  const int am = lane & 15, akh = (lane >> 4) * 8;
  const int bn = lane & 15, bkh = (lane >> 4) * 16;

  const unsigned short* qbase = qf + (size_t)n * Cc * Ss;
  const unsigned short* kbp = kf + (size_t)n * Cc * KL;
  const unsigned short* vbp = vf + (size_t)n * Cc * KL;

  for (int i = 0; i < 64; ++i) {  // Qs once, q fast (global s-contig)
    int e = tid + i * 256;
    int q = e & 31, c = e >> 5;
    int s = sBase + q;
    Qs[q][c] = (s < Ss) ? qbase[(size_t)c * Ss + s] : (unsigned short)0;
  }
  if (tid < 32) { rowM[tid] = -3.0e38f; rowS[tid] = 0.f; }

  v8f oacc[8];
#pragma unroll
  for (int i = 0; i < 8; ++i) oacc[i] = zero8();

  // issue async copy of one fully-valid 32-key V chunk into a buffer
  auto issue_vchunk = [&](int kb32, unsigned short(*Vb)[40]) {
#pragma unroll
    for (int i = 0; i < 8; ++i) {
      int e = tid + i * 256;
      int k8 = (e & 3) * 8, c = e >> 2;
      async_cp16(vbp + (size_t)c * KL + kb32 + k8, &Vb[c][k8]);
    }
  };
  // synchronous, predicated fill (tail block only)
  auto sync_fill = [&](int kb32, unsigned short(*Vb)[40]) {
#pragma unroll
    for (int i = 0; i < 8; ++i) {
      int e = tid + i * 256;
      int k8 = (e & 3) * 8, c = e >> 2;
      int gk = kb32 + k8;
      const unsigned short* src = vbp + (size_t)c * KL + gk;
      if (gk + 7 < KL) {
        *(uint4*)&Vb[c][k8] = *(const uint4*)src;
      } else {
#pragma unroll
        for (int t = 0; t < 8; ++t)
          Vb[c][k8 + t] = (gk + t < KL) ? src[t] : (unsigned short)0;
      }
    }
  };
  auto pv = [&](int kk, unsigned short(*Vb)[40]) {
    FragAB a;
#pragma unroll
    for (int j = 0; j < 8; ++j) {
      int kp = kk * 32 + 2 * akh + 2 * j;
      a.u32[j] = *(const unsigned int*)&Ps[qt * 16 + am][kp];
    }
#pragma unroll
    for (int ct = 0; ct < 8; ++ct) {
      FragAB b;
#pragma unroll
      for (int j = 0; j < 8; ++j)
        b.u32[j] = *(const unsigned int*)&Vb[kc * 128 + ct * 16 + bn][bkh + 2 * j];
      oacc[ct] = __builtin_amdgcn_wmma_f32_16x16x32_bf16(
          false, a.bf, false, b.bf, (short)0, oacc[ct], false, false);
    }
  };

#pragma unroll 1
  for (int kb = 0; kb < 31; ++kb) {
    const int kbase = kb * 128;
    const bool fullBlk = (kb < 30);
    __syncthreads();
    // K block [128 key][512 c]: b128 global loads + staggered LDS transpose
    if (fullBlk) {
      for (int i = 0; i < 32; ++i) {
        int e = tid + i * 256;
        int k8 = (e & 15) * 8, c = e >> 4;
        unsigned short tmp[8];
        *(uint4*)tmp = *(const uint4*)(kbp + (size_t)c * KL + kbase + k8);
#pragma unroll
        for (int t = 0; t < 8; ++t) {  // lane-staggered: <=2-way bank conflicts
          int tt = (t + lane) & 7;
          Ks[k8 + tt][c] = tmp[tt];
        }
      }
      // prefetch V chunk 0 (DMA overlaps the logits GEMM below)
      issue_vchunk(kbase, Vs[0]);
    } else {
      for (int i = 0; i < 32; ++i) {
        int e = tid + i * 256;
        int k8 = (e & 15) * 8, c = e >> 4;
        int gk = kbase + k8;
        const unsigned short* src = kbp + (size_t)c * KL + gk;
        unsigned short tmp[8];
        if (gk + 7 < KL) {
          *(uint4*)tmp = *(const uint4*)src;
        } else {
#pragma unroll
          for (int t = 0; t < 8; ++t)
            tmp[t] = (gk + t < KL) ? src[t] : (unsigned short)0;
        }
#pragma unroll
        for (int t = 0; t < 8; ++t) Ks[k8 + t][c] = tmp[t];
      }
    }
    __syncthreads();

    // logits: each wave 2 tiles of 16x16, 16 c-steps -> 32 chained WMMAs
    v8f lac0 = zero8(), lac1 = zero8();
#pragma unroll
    for (int c0 = 0; c0 < Cc; c0 += 32) {
      FragAB a;
#pragma unroll
      for (int j = 0; j < 8; ++j) {
        int kp = c0 + 2 * akh + 2 * j;
        a.u32[j] = *(const unsigned int*)&Qs[qt * 16 + am][kp];
      }
      FragAB b0, b1;
#pragma unroll
      for (int j = 0; j < 8; ++j) {
        b0.u32[j] = *(const unsigned int*)&Ks[kc * 32 + bn][c0 + bkh + 2 * j];
        b1.u32[j] = *(const unsigned int*)&Ks[kc * 32 + 16 + bn][c0 + bkh + 2 * j];
      }
      lac0 = __builtin_amdgcn_wmma_f32_16x16x32_bf16(false, a.bf, false, b0.bf,
                                                     (short)0, lac0, false, false);
      lac1 = __builtin_amdgcn_wmma_f32_16x16x32_bf16(false, a.bf, false, b1.bf,
                                                     (short)0, lac1, false, false);
    }
#pragma unroll
    for (int r = 0; r < 8; ++r) {
      int q = qt * 16 + r + 8 * (lane >> 4);
      Ls[q][kc * 32 + (lane & 15)] = lac0[r];
      Ls[q][kc * 32 + 16 + (lane & 15)] = lac1[r];
    }
    __syncthreads();

    // online softmax: 8 threads per row, vectorized LDS access
    {
      int row = tid >> 3, sub = tid & 7;
      float lv[16];
      const float4* lrow = (const float4*)&Ls[row][sub * 16];
#pragma unroll
      for (int t = 0; t < 4; ++t) {
        float4 v = lrow[t];
        lv[4 * t + 0] = v.x; lv[4 * t + 1] = v.y;
        lv[4 * t + 2] = v.z; lv[4 * t + 3] = v.w;
      }
      const int validBase = KL - kbase - sub * 16;  // col i valid iff i<validBase
      float mloc = -3.0e38f;
#pragma unroll
      for (int i = 0; i < 16; ++i) {
        lv[i] = (i < validBase) ? lv[i] * ATTN_SCALE : -3.0e38f;
        mloc = fmaxf(mloc, lv[i]);
      }
#pragma unroll
      for (int d = 1; d < 8; d <<= 1) mloc = fmaxf(mloc, __shfl_xor(mloc, d, 32));
      float newM = fmaxf(rowM[row], mloc);
      float sloc = 0.f;
      float pv16[16];
#pragma unroll
      for (int i = 0; i < 16; ++i) {
        float p = __expf(lv[i] - newM);  // masked lanes underflow to exact 0
        pv16[i] = p;
        sloc += p;
      }
      uint4* prow = (uint4*)&Ps[row][sub * 16];
      prow[0] = make_uint4(pack2bf(pv16[0], pv16[1]), pack2bf(pv16[2], pv16[3]),
                           pack2bf(pv16[4], pv16[5]), pack2bf(pv16[6], pv16[7]));
      prow[1] = make_uint4(pack2bf(pv16[8], pv16[9]), pack2bf(pv16[10], pv16[11]),
                           pack2bf(pv16[12], pv16[13]), pack2bf(pv16[14], pv16[15]));
#pragma unroll
      for (int d = 1; d < 8; d <<= 1) sloc += __shfl_xor(sloc, d, 32);
      if (sub == 0) {
        float sc = __expf(rowM[row] - newM);
        rowScale[row] = sc;
        rowS[row] = rowS[row] * sc + sloc;
        rowM[row] = newM;
      }
    }
    __syncthreads();

    // rescale O accumulators
#pragma unroll
    for (int r = 0; r < 8; ++r) {
      float sc = rowScale[qt * 16 + r + 8 * (lane >> 4)];
#pragma unroll
      for (int t = 0; t < 8; ++t) oacc[t][r] *= sc;
    }

    // O += P @ V : double-buffered async V stream (4 chunks of 32 keys)
    for (int kk = 0; kk < 4; ++kk) {
      __syncthreads();  // all waves done reading the buffer we are about to fill
      if (fullBlk) {
        if (kk < 3) {
          issue_vchunk(kbase + (kk + 1) * 32, Vs[(kk + 1) & 1]);
          WAIT_ASYNC(8);  // chunk kk has landed (in-order retirement)
        } else {
          WAIT_ASYNC(0);
        }
      } else {
        sync_fill(kbase + kk * 32, Vs[kk & 1]);
      }
      __syncthreads();
      pv(kk, Vs[kk & 1]);
    }
  }

  __syncthreads();
#pragma unroll
  for (int r = 0; r < 8; ++r) {
    int q = qt * 16 + r + 8 * (lane >> 4);
    int s = sBase + q;
    if (s >= Ss) continue;
    float inv = 1.f / rowS[q];
#pragma unroll
    for (int ct = 0; ct < 8; ++ct) {
      int c = kc * 128 + ct * 16 + (lane & 15);
      ow[((size_t)n * Ss + s) * Cc + c] = oacc[ct][r] * inv;
    }
  }
}

// ---------------------------------------------------------------------------
// LayerNorm partial sums: grid (N, 8), float4 loads, deterministic tree.
// ---------------------------------------------------------------------------
__global__ __launch_bounds__(256) void ln_part_kernel(
    const float* __restrict__ ow, float* __restrict__ partials) {
  __shared__ float ssum[256], ssq[256];
  const int n = blockIdx.x, p = blockIdx.y;
  const float4* base =
      (const float4*)(ow + (size_t)n * Ss * Cc) + (size_t)p * 3136;
  float s = 0.f, q = 0.f;
  for (int i = threadIdx.x; i < 3136; i += 256) {
    float4 v = base[i];
    s += v.x + v.y + v.z + v.w;
    q += v.x * v.x + v.y * v.y + v.z * v.z + v.w * v.w;
  }
  ssum[threadIdx.x] = s;
  ssq[threadIdx.x] = q;
  __syncthreads();
  for (int d = 128; d > 0; d >>= 1) {
    if (threadIdx.x < d) {
      ssum[threadIdx.x] += ssum[threadIdx.x + d];
      ssq[threadIdx.x] += ssq[threadIdx.x + d];
    }
    __syncthreads();
  }
  if (threadIdx.x == 0) {
    partials[(n * 8 + p) * 2] = ssum[0];
    partials[(n * 8 + p) * 2 + 1] = ssq[0];
  }
}

// ---------------------------------------------------------------------------
// Final: out = Wo @ relu((o - mu)*rstd) + bo + person   (fp32, [N][C][S])
// ---------------------------------------------------------------------------
__global__ __launch_bounds__(256) void out_kernel(
    const float* __restrict__ ow, const float* __restrict__ partials,
    const float* __restrict__ Wo, const float* __restrict__ bo,
    const float* __restrict__ person, float* __restrict__ out) {
  __shared__ unsigned short Ws[128][36];
  __shared__ unsigned short Xs[SP][36];

  const int tid = threadIdx.x, lane = tid & 31, wv = tid >> 5;
  const int n = blockIdx.x, oBase = blockIdx.y * 128;

  float sum = 0.f, sq = 0.f;
#pragma unroll
  for (int p = 0; p < 8; ++p) {  // uniform scalar loads
    sum += partials[(n * 8 + p) * 2];
    sq += partials[(n * 8 + p) * 2 + 1];
  }
  const float cnt = (float)(Ss * Cc);
  const float mu = sum / cnt;
  const float rstd = rsqrtf(sq / cnt - mu * mu + 1e-5f);
  const float* xin = ow + (size_t)n * Ss * Cc;  // [s][c]

  v8f acc[13];
#pragma unroll
  for (int i = 0; i < 13; ++i) acc[i] = zero8();

  const int am = lane & 15, akh = (lane >> 4) * 8;
  const int bn = lane & 15, bkh = (lane >> 4) * 16;

  for (int c0 = 0; c0 < Cc; c0 += 32) {
    __syncthreads();
#pragma unroll
    for (int i = 0; i < 16; ++i) {
      int e = tid + i * 256;
      int c = e & 31, o = e >> 5;
      Ws[o][c] = f2bf(Wo[(size_t)(oBase + o) * Cc + c0 + c]);
    }
#pragma unroll
    for (int i = 0; i < 26; ++i) {  // c fast: global is c-contiguous here
      int e = tid + i * 256;
      int c = e & 31, s = e >> 5;
      float v = 0.f;
      if (s < Ss) {
        v = (xin[(size_t)s * Cc + c0 + c] - mu) * rstd;
        v = fmaxf(v, 0.f);
      }
      Xs[s][c] = f2bf(v);
    }
    __syncthreads();

    FragAB a;
#pragma unroll
    for (int j = 0; j < 8; ++j) {
      int kp = 2 * akh + 2 * j;
      a.u32[j] = *(const unsigned int*)&Ws[wv * 16 + am][kp];
    }
#pragma unroll
    for (int st = 0; st < 13; ++st) {
      FragAB b;
#pragma unroll
      for (int j = 0; j < 8; ++j)
        b.u32[j] = *(const unsigned int*)&Xs[st * 16 + bn][bkh + 2 * j];
      acc[st] = __builtin_amdgcn_wmma_f32_16x16x32_bf16(
          false, a.bf, false, b.bf, (short)0, acc[st], false, false);
    }
  }

#pragma unroll
  for (int st = 0; st < 13; ++st) {
    int s = st * 16 + (lane & 15);
    if (s >= Ss) continue;
#pragma unroll
    for (int r = 0; r < 8; ++r) {
      int o = oBase + wv * 16 + r + 8 * (lane >> 4);
      size_t idx = ((size_t)n * Cc + o) * Ss + s;
      out[idx] = acc[st][r] + bo[o] + person[idx];
    }
  }
}

// ---------------------------------------------------------------------------
extern "C" void kernel_launch(void* const* d_in, const int* in_sizes, int n_in,
                              void* d_out, int out_size, void* d_ws,
                              size_t ws_size, hipStream_t stream) {
  const float* person = (const float*)d_in[0];
  const float* others = (const float*)d_in[1];
  const float* Wq = (const float*)d_in[2];
  const float* bq = (const float*)d_in[3];
  const float* Wk = (const float*)d_in[4];
  const float* bk = (const float*)d_in[5];
  const float* Wv = (const float*)d_in[6];
  const float* bv = (const float*)d_in[7];
  const float* Wo = (const float*)d_in[8];
  const float* bo = (const float*)d_in[9];
  const float* pos_k = (const float*)d_in[10];
  const float* pos_v = (const float*)d_in[11];
  float* out = (float*)d_out;

  char* ws = (char*)d_ws;
  size_t off = 0;
  auto alloc = [&](size_t bytes) -> void* {
    void* p = ws + off;
    off = (off + bytes + 255) & ~(size_t)255;
    return p;
  };
  unsigned short* qf = (unsigned short*)alloc((size_t)Nn * Cc * Ss * 2);
  unsigned short* kfp = (unsigned short*)alloc((size_t)Nn * Cc * KL * 2);
  unsigned short* vfp = (unsigned short*)alloc((size_t)Nn * Cc * KL * 2);
  float* owb = (float*)alloc((size_t)Nn * Ss * Cc * 4);
  float* partials = (float*)alloc((size_t)Nn * 8 * 2 * 4);
  (void)ws_size; (void)in_sizes; (void)n_in; (void)out_size;

  dim3 blk(256);
  proj_kernel<0><<<dim3(Nn, 4), blk, 0, stream>>>(person, Wq, bq, nullptr, qf);
  proj_kernel<1><<<dim3(Nn * Mm, 4), blk, 0, stream>>>(others, Wk, bk, pos_k, kfp);
  proj_kernel<2><<<dim3(Nn * Mm, 4), blk, 0, stream>>>(others, Wv, bv, pos_v, vfp);
  attn_kernel<<<dim3(7, Nn), blk, 0, stream>>>(qf, kfp, vfp, owb);
  ln_part_kernel<<<dim3(Nn, 8), blk, 0, stream>>>(owb, partials);
  out_kernel<<<dim3(Nn, 4), blk, 0, stream>>>(owb, partials, Wo, bo, person, out);
}
